// LIF_63221918597752
// MI455X (gfx1250) — matrix operand order repaired and checked by reading, not compiled, not run on hardware
//
#include <hip/hip_runtime.h>
#include <hip/hip_bf16.h>

// Problem constants (from the reference)
#define BB   64
#define TTT  1000
#define DIN  256
#define NNN  512
#define DECAY 0.95122942450071400f   // exp(-1/20)
#define SCAN_BLOCKS 16               // 16 blocks * 8 waves = 128 tiles (exact)

typedef __attribute__((ext_vector_type(16))) __bf16 v16bf;
typedef __attribute__((ext_vector_type(8)))  float  v8f;

// Build the A-operand lane vector for v_wmma_f32_16x16x32_bf16 from an f32 row.
// Contiguous K layout: elements 0..15 <- K = kA..kA+15
// (kA already includes the lane-half shift of 16).
__device__ __forceinline__ v16bf load_a_bf16(const float* __restrict__ row, int kA) {
    const float4* p0 = reinterpret_cast<const float4*>(row + kA);
    const float4* p1 = reinterpret_cast<const float4*>(row + kA + 8);
    float4 f0 = p0[0], f1 = p0[1], f2 = p1[0], f3 = p1[1];
    v16bf a;
    a[0]  = (__bf16)f0.x; a[1]  = (__bf16)f0.y; a[2]  = (__bf16)f0.z; a[3]  = (__bf16)f0.w;
    a[4]  = (__bf16)f1.x; a[5]  = (__bf16)f1.y; a[6]  = (__bf16)f1.z; a[7]  = (__bf16)f1.w;
    a[8]  = (__bf16)f2.x; a[9]  = (__bf16)f2.y; a[10] = (__bf16)f2.z; a[11] = (__bf16)f2.w;
    a[12] = (__bf16)f3.x; a[13] = (__bf16)f3.y; a[14] = (__bf16)f3.z; a[15] = (__bf16)f3.w;
    return a;
}

// One-shot prep: convert W, R to bf16; zero t=0 state slices; reset the barrier counter.
__global__ void lif_prep(const float* __restrict__ W, const float* __restrict__ R,
                         __bf16* __restrict__ Wbf, __bf16* __restrict__ Rbf,
                         float* __restrict__ u_all, float* __restrict__ z_all,
                         unsigned* __restrict__ counter) {
    int idx = blockIdx.x * blockDim.x + threadIdx.x;
    if (idx == 0) counter[0] = 0u;
    if (idx < NNN * DIN) Wbf[idx] = (__bf16)W[idx];
    if (idx < NNN * NNN) Rbf[idx] = (__bf16)R[idx];
    if (idx < BB * NNN) {
        int b = idx / NNN, n = idx % NNN;
        size_t off = (size_t)b * (TTT + 1) * NNN + n;   // time slice 0
        u_all[off] = 0.0f;
        z_all[off] = 0.0f;
    }
}

// Feed-forward GEMM: incur[m, n] = sum_k x[m,k] * W[n,k] + bias[n]
// M = B*T = 64000 (4000 tiles), N = 512. Each wave: 1 m-tile x 4 n-tiles
// (one A load feeds 4 WMMAs -> 4x less A traffic).
__global__ void lif_ff_gemm(const float* __restrict__ x, const __bf16* __restrict__ Wbf,
                            const float* __restrict__ bias, float* __restrict__ incur) {
    const int lane = threadIdx.x & 31;
    const int wave = blockIdx.x * (blockDim.x >> 5) + (threadIdx.x >> 5);
    const int mt = wave >> 3;          // 4000 m-tiles
    const int ng = wave & 7;           // 8 groups of 4 n-tiles
    const int m0 = mt * 16, n0 = ng * 64;

    const int arow  = m0 + (lane & 15);
    const int kAoff = (lane & 16) ? 16 : 0;
    const int kBoff = (lane & 16) ? 16 : 0;

    const float* xr = x + (size_t)arow * DIN;
    const __bf16* wr[4];
    int bn[4];
#pragma unroll
    for (int j = 0; j < 4; ++j) {
        bn[j] = n0 + j * 16 + (lane & 15);
        wr[j] = Wbf + (size_t)bn[j] * DIN;
    }

    v8f acc[4] = {};
#pragma unroll
    for (int k0 = 0; k0 < DIN; k0 += 32) {
        v16bf a = load_a_bf16(xr, k0 + kAoff);
#pragma unroll
        for (int j = 0; j < 4; ++j) {
            v16bf b = *reinterpret_cast<const v16bf*>(wr[j] + k0 + kBoff);
            acc[j] = __builtin_amdgcn_wmma_f32_16x16x32_bf16(
                         false, a, false, b, (short)0, acc[j], false, false);
        }
    }

#pragma unroll
    for (int j = 0; j < 4; ++j) {
        const float bv = bias[bn[j]];
#pragma unroll
        for (int r = 0; r < 8; ++r) {
            int m = m0 + r + ((lane & 16) ? 8 : 0);
            incur[(size_t)m * NNN + bn[j]] = acc[j][r] + bv;
        }
    }
}

// Persistent LIF scan: 16 blocks x 8 waves = 128 tiles of the 64x512 state.
// Each wave: preloads its 16 K-chunks of R into registers once, keeps its
// u/z output tile in registers across all T steps, and syncs the grid with a
// device-scope atomic counter + acquire/release fences between steps.
__global__ void lif_scan(const float* __restrict__ incur, const __bf16* __restrict__ Rbf,
                         float* __restrict__ outp, float* __restrict__ u_all,
                         float* __restrict__ z_all, unsigned* __restrict__ counter) {
    const int lane = threadIdx.x & 31;
    const int warp = threadIdx.x >> 5;
    const int w  = blockIdx.x * 8 + warp;
    const int mt = w >> 5;             // 4 batch tiles
    const int nt = w & 31;             // 32 neuron tiles
    const int m0 = mt * 16, n0 = nt * 16;

    const int arow  = m0 + (lane & 15);
    const int bn    = n0 + (lane & 15);
    const int kAoff = (lane & 16) ? 16 : 0;
    const int kBoff = (lane & 16) ? 16 : 0;

    // Preload this wave's R columns (time-invariant B operand) into registers.
    const __bf16* rr = Rbf + (size_t)bn * NNN;
    v16bf rB[16];
#pragma unroll
    for (int c = 0; c < 16; ++c)
        rB[c] = *reinterpret_cast<const v16bf*>(rr + c * 32 + kBoff);

    const size_t st = (size_t)(TTT + 1) * NNN;          // per-batch state stride
    const int rowhi = (lane & 16) ? 8 : 0;

    float ureg[8], zreg[8];
#pragma unroll
    for (int r = 0; r < 8; ++r) { ureg[r] = 0.0f; zreg[r] = 0.0f; }

#pragma unroll 1
    for (int t = 1; t <= TTT; ++t) {
        const float* zp_row = z_all + (size_t)arow * st + (size_t)(t - 1) * NNN;

        v8f acc = {};
#pragma unroll
        for (int c = 0; c < 16; ++c) {
            v16bf a = load_a_bf16(zp_row, c * 32 + kAoff);   // z in {0,1}: exact in bf16
            acc = __builtin_amdgcn_wmma_f32_16x16x32_bf16(
                      false, a, false, rB[c], (short)0, acc, false, false);
        }

#pragma unroll
        for (int r = 0; r < 8; ++r) {
            int bidx   = m0 + r + rowhi;
            size_t cur = (size_t)bidx * st + (size_t)t * NNN + bn;
            size_t seq = (size_t)bidx * TTT * NNN + (size_t)(t - 1) * NNN + bn;

            float soma = acc[r] + incur[seq];
            float u    = ureg[r] * (1.0f - zreg[r]);        // reset with detached spike
            float ut   = DECAY * u + (1.0f - DECAY) * soma;
            float zt   = (ut - 1.0f > 0.0f) ? 1.0f : 0.0f;  // heaviside(u - THR)

            ureg[r] = ut;
            zreg[r] = zt;
            u_all[cur] = ut;
            z_all[cur] = zt;
            outp[seq]  = zt;
        }

        // Grid-wide barrier: make this step's z stores visible to all WGPs.
        __builtin_amdgcn_fence(__ATOMIC_RELEASE, "agent");  // per-wave store flush
        __syncthreads();
        if (threadIdx.x == 0) {
            __hip_atomic_fetch_add(counter, 1u, __ATOMIC_RELAXED, __HIP_MEMORY_SCOPE_AGENT);
            const unsigned target = (unsigned)t * SCAN_BLOCKS;
            while (__hip_atomic_load(counter, __ATOMIC_RELAXED, __HIP_MEMORY_SCOPE_AGENT) < target)
                __builtin_amdgcn_s_sleep(1);
        }
        __syncthreads();
        __builtin_amdgcn_fence(__ATOMIC_ACQUIRE, "agent"); // invalidate read path
    }
}

extern "C" void kernel_launch(void* const* d_in, const int* in_sizes, int n_in,
                              void* d_out, int out_size, void* d_ws, size_t ws_size,
                              hipStream_t stream) {
    const float* x    = (const float*)d_in[0];   // [B,T,Din]
    const float* W    = (const float*)d_in[1];   // [N,Din]
    const float* bias = (const float*)d_in[2];   // [N]
    const float* R    = (const float*)d_in[3];   // [N,N]

    // d_out = outputs [B,T,N] ++ u_all [B,T+1,N] ++ z_all [B,T+1,N]
    float* outp  = (float*)d_out;
    float* u_all = outp  + (size_t)BB * TTT * NNN;
    float* z_all = u_all + (size_t)BB * (TTT + 1) * NNN;

    // d_ws = incur f32 [B*T, N] ++ Wbf bf16 [N,Din] ++ Rbf bf16 [N,N] ++ counter
    float*    incur   = (float*)d_ws;
    __bf16*   Wbf     = (__bf16*)(incur + (size_t)BB * TTT * NNN);
    __bf16*   Rbf     = Wbf + (size_t)NNN * DIN;
    unsigned* counter = (unsigned*)(Rbf + (size_t)NNN * NNN);

    lif_prep<<<(NNN * NNN + 255) / 256, 256, 0, stream>>>(W, R, Wbf, Rbf, u_all, z_all, counter);

    // 4000 m-tiles * 8 n-groups = 32000 waves, 8 waves/block -> 4000 blocks (exact)
    lif_ff_gemm<<<4000, 256, 0, stream>>>(x, Wbf, bias, incur);

    // Persistent scan: 16 co-resident blocks, internal grid barrier per step
    lif_scan<<<SCAN_BLOCKS, 256, 0, stream>>>(incur, Rbf, outp, u_all, z_all, counter);
}
